// WassersteinLoss_27556510171553
// MI455X (gfx1250) — matrix operand order repaired and checked by reading, not compiled, not run on hardware
//
#include <hip/hip_runtime.h>

typedef _Float16 v16h __attribute__((ext_vector_type(16)));
typedef float    v8f  __attribute__((ext_vector_type(8)));
typedef float    v4f  __attribute__((ext_vector_type(4)));

static constexpr int IMG_H  = 512;
static constexpr int IMG_W  = 512;
static constexpr int NCH    = 128;   // B*C channels per tensor
static constexpr int STRIPS = IMG_H / 16;   // 32 sixteen-row strips
static constexpr int KSTEPS = IMG_W / 32;   // 16 WMMA K-steps per strip
static constexpr int WAVES  = 16;           // waves per block
static constexpr int TPB    = WAVES * 32;   // 512 threads

// ---------------------------------------------------------------------------
// Kernel 1: per-(tensor,channel) nonzero-coordinate moments via WMMA.
//   A (16x32 f16)  = 0/1 mask tile of 16 image rows x 32 columns
//   B (32x16 f16)  = column features per k: {1, c, ch^2, ch*cl, cl^2}, c=32ch+cl
//   C (16x16 f32) += A x B   -> per-row sums of each feature (exact: all
//                               f16 operands are integers < 2048)
// Row weights r, r^2 folded in per strip on the f32 accumulator.
// Output: ws[block*6 + {N, Sr, Srr, Sc, Src, Scc}]
// ---------------------------------------------------------------------------
__global__ __launch_bounds__(TPB)
void moments_kernel(const float* __restrict__ pred,
                    const float* __restrict__ target,
                    float* __restrict__ ws)
{
    __shared__ alignas(32) _Float16 Bld[KSTEPS * 32 * 16];   // 16 KB feature matrix
    __shared__ float stats[6];

    const int tid  = threadIdx.x;
    const int lane = tid & 31;
    const int wave = tid >> 5;

    // ---- precompute B in LDS (identical for every channel / strip) ----
    for (int e = tid; e < KSTEPS * 32 * 8; e += TPB) {
        const int v    = e & 7;           // VGPR pair within B
        const int bl   = (e >> 3) & 31;   // lane of B layout
        const int step = e >> 8;          // K-slice
        const int j    = bl & 15;         // N column = feature index
        const int koff = (bl < 16) ? 0 : 8;
        #pragma unroll
        for (int hbit = 0; hbit < 2; ++hbit) {
            const int c  = step * 32 + koff + 2 * v + hbit + ((v & 4) ? 8 : 0);  // absolute column
            const int ch = c >> 5, cl = c & 31;
            float f;
            switch (j) {
                case 0:  f = 1.0f;                 break;   // count
                case 1:  f = (float)c;             break;   // Sc, Src
                case 2:  f = (float)(ch * ch);     break;   // Scc (x1024)
                case 3:  f = (float)(ch * cl);     break;   // Scc (x64)
                case 4:  f = (float)(cl * cl);     break;   // Scc (x1)
                default: f = 0.0f;                 break;
            }
            Bld[(step * 32 + bl) * 16 + 2 * v + hbit] = (_Float16)f;
        }
    }
    if (tid < 6) stats[tid] = 0.0f;
    __syncthreads();

    // ---- image for this block ----
    const int chan = (int)(blockIdx.x & (NCH - 1));
    const float* src = ((int)blockIdx.x < NCH ? pred : target)
                       + (size_t)chan * IMG_H * IMG_W;

    // 16-bit A layout: lanes 0-15 -> row=lane, K in {0..7,16..23};
    //                  lanes 16-31 -> row=lane-16, K in {8..15,24..31}
    const int rowInStrip = lane & 15;
    const int koffA = (lane < 16) ? 0 : 8;
    const int mOff  = (lane < 16) ? 0 : 8;   // C layout: lanes>=16 hold rows M+8

    float t0 = 0.f, t1 = 0.f, t2 = 0.f;      // Σf, Σ r·f, Σ r²·f  (per feature)

    for (int s = wave; s < STRIPS; s += WAVES) {
        const int rbase = s * 16;
        const float* rowp = src + (size_t)(rbase + rowInStrip) * IMG_W + koffA;

        // prefetch next strip this wave will touch (speculative)
        {
            int sn = (s + WAVES < STRIPS) ? (s + WAVES) : s;
            __builtin_prefetch(src + (size_t)(sn * 16 + rowInStrip) * IMG_W + koffA, 0, 1);
        }

        v8f acc = {};
        #pragma unroll 2
        for (int step = 0; step < KSTEPS; ++step) {
            const float* p = rowp + step * 32;
            v4f x0 = *(const v4f*)(p);
            v4f x1 = *(const v4f*)(p + 4);
            v4f x2 = *(const v4f*)(p + 16);
            v4f x3 = *(const v4f*)(p + 20);

            v16h a;
            #pragma unroll
            for (int i = 0; i < 4; ++i) {
                a[i]      = (x0[i] != 0.f) ? (_Float16)1.0f : (_Float16)0.0f;
                a[i + 4]  = (x1[i] != 0.f) ? (_Float16)1.0f : (_Float16)0.0f;
                a[i + 8]  = (x2[i] != 0.f) ? (_Float16)1.0f : (_Float16)0.0f;
                a[i + 12] = (x3[i] != 0.f) ? (_Float16)1.0f : (_Float16)0.0f;
            }
            v16h b = *(const v16h*)&Bld[(step * 32 + lane) * 16];

            acc = __builtin_amdgcn_wmma_f32_16x16x32_f16(
                      false, a, false, b, (short)0, acc, false, false);
        }

        // fold row weights: lane's VGPR v holds row M = v + mOff of this strip
        #pragma unroll
        for (int v = 0; v < 8; ++v) {
            const float r   = (float)(rbase + v + mOff);
            const float val = acc[v];
            t0 += val;
            t1 += r * val;
            t2 += r * r * val;
        }
    }

    // ---- combine across lanes/waves via LDS float atomics (ds_add_f32) ----
    const int j = lane & 15;                 // feature this lane accumulated
    if (j == 0) {
        atomicAdd(&stats[0], t0);            // N
        atomicAdd(&stats[1], t1);            // Sr
        atomicAdd(&stats[2], t2);            // Srr
    } else if (j == 1) {
        atomicAdd(&stats[3], t0);            // Sc
        atomicAdd(&stats[4], t1);            // Src
    } else if (j == 2) {
        atomicAdd(&stats[5], 1024.0f * t0);  // Scc: 1024*Σw*ch²
    } else if (j == 3) {
        atomicAdd(&stats[5], 64.0f * t0);    //       +64*Σw*ch*cl
    } else if (j == 4) {
        atomicAdd(&stats[5], t0);            //       +Σw*cl²
    }
    __syncthreads();

    if (tid < 6) ws[(size_t)blockIdx.x * 6 + tid] = stats[tid];
}

// ---------------------------------------------------------------------------
// Kernel 2: closed-form 2-Wasserstein per channel + mean over 128 channels.
// ---------------------------------------------------------------------------
__global__ __launch_bounds__(128)
void wasserstein_kernel(const float* __restrict__ ws, float* __restrict__ out)
{
    __shared__ float red[128];
    const int ch = threadIdx.x;
    const float EPS = 1e-8f;
    const float* s1 = ws + (size_t)ch * 6;
    const float* s2 = ws + (size_t)(NCH + ch) * 6;

    auto gauss = [&](const float* s, float& mr, float& mc,
                     float& vrr, float& vcc, float& vrc) {
        float N = s[0], Sr = s[1], Srr = s[2], Sc = s[3], Src = s[4], Scc = s[5];
        float Ns  = fmaxf(N, 1.f);
        mr = Sr / Ns;  mc = Sc / Ns;
        float den = fmaxf(N - 1.f, 1.f);
        vrr = (Srr - N * mr * mr) / den + EPS;
        vcc = (Scc - N * mc * mc) / den + EPS;
        vrc = (Src - N * mr * mc) / den;
    };

    float mr1, mc1, vrr1, vcc1, vrc1, mr2, mc2, vrr2, vcc2, vrc2;
    gauss(s1, mr1, mc1, vrr1, vcc1, vrc1);
    gauss(s2, mr2, mc2, vrr2, vcc2, vrc2);

    float term1 = (mr1 - mr2) * (mr1 - mr2) + (mc1 - mc2) * (mc1 - mc2);
    float tr1   = vrr1 + vcc1, tr2 = vrr2 + vcc2;
    float det1  = vrr1 * vcc1 - vrc1 * vrc1;
    float det2  = vrr2 * vcc2 - vrc2 * vrc2;
    float tr12  = vrr1 * vrr2 + 2.f * vrc1 * vrc2 + vcc1 * vcc2;
    float cross = sqrtf(fmaxf(tr12 + 2.f * sqrtf(fmaxf(det1 * det2, 0.f)), 0.f));
    float term2 = tr1 + tr2 - 2.f * cross - 4.f * EPS;
    float dist  = sqrtf(fmaxf(term1 + term2 + EPS, 0.f));

    red[ch] = dist;
    __syncthreads();
    #pragma unroll
    for (int off = 64; off > 0; off >>= 1) {
        if (ch < off) red[ch] += red[ch + off];
        __syncthreads();
    }
    if (ch == 0) out[0] = red[0] * (1.0f / NCH);
}

// ---------------------------------------------------------------------------
extern "C" void kernel_launch(void* const* d_in, const int* in_sizes, int n_in,
                              void* d_out, int out_size, void* d_ws, size_t ws_size,
                              hipStream_t stream) {
    (void)in_sizes; (void)n_in; (void)out_size; (void)ws_size;
    const float* pred   = (const float*)d_in[0];
    const float* target = (const float*)d_in[1];
    float* ws  = (float*)d_ws;    // 2*NCH*6 floats = 6 KB, fully rewritten each call
    float* out = (float*)d_out;   // one float

    moments_kernel<<<2 * NCH, TPB, 0, stream>>>(pred, target, ws);
    wasserstein_kernel<<<1, 128, 0, stream>>>(ws, out);
}
